// DeltaNetLayer_11244224381580
// MI455X (gfx1250) — matrix operand order repaired and checked by reading, not compiled, not run on hardware
//
#include <hip/hip_runtime.h>

// DeltaNet layer for MI455X (gfx1250, wave32, WMMA 16x16x32 bf16).
// Pipeline: f32->bf16 cvt -> 3x WMMA proj GEMMs (async-LDS double-buffered)
// -> beta proj -> depthwise conv(K=4) -> per-head norms/silu (+ relayout to
// [B,H,L,HD]) -> chunked WY delta-rule scan (C=64, all matmuls on WMMA, S kept
// in accumulator registers, async chunk loads) -> RMSNorm -> WMMA output GEMM.

#define DEV static __device__ __forceinline__

#define Bb 8
#define Ls 2048
#define Dd 1024
#define Hh 16
#define HD 64
#define BL (Bb * Ls)          // 16384
#define CHUNK 64
#define NCHUNK (Ls / CHUNK)   // 32

typedef __bf16 bf16_t;
typedef bf16_t v16bf __attribute__((ext_vector_type(16)));
typedef float  v8f   __attribute__((ext_vector_type(8)));

struct alignas(16) U4 { unsigned int x, y, z, w; };
union FragBF { v16bf v; U4 q[2]; };

DEV unsigned short f2bf(float f) {
  unsigned int u = __float_as_uint(f);
  return (unsigned short)((u + 0x7FFFu + ((u >> 16) & 1u)) >> 16);   // RNE
}
DEV float bf2f(unsigned short h) { return __uint_as_float(((unsigned int)h) << 16); }

DEV v8f wmma_bf16(const FragBF& a, const FragBF& b, v8f c) {
  return __builtin_amdgcn_wmma_f32_16x16x32_bf16(false, a.v, false, b.v, (short)0, c, false, false);
}

// ---- CDNA5 async global->LDS copy (16B per lane), tracked by ASYNCcnt ----
DEV void async_copy_b128(const void* gptr, void* ldsptr) {
  unsigned laddr = (unsigned)(size_t)ldsptr;   // generic LDS addr truncates to LDS offset
  asm volatile("global_load_async_to_lds_b128 %0, %1, off"
               :: "v"(laddr), "v"(gptr) : "memory");
}
DEV void wait_async0() {
#if __has_builtin(__builtin_amdgcn_s_wait_asynccnt)
  __builtin_amdgcn_s_wait_asynccnt(0);
#else
  asm volatile("s_wait_asynccnt 0x0" ::: "memory");
#endif
}

// A fragment (16x32 bf16) from row-major [M,K]: lanes 0-15 row m0+lane hold
// K {k0..k0+7, k0+16..k0+23}; lanes 16-31 hold {k0+8..15, k0+24..31}.
DEV void load_frag_a(FragBF& f, const unsigned short* base, int row0, int k0, int ld, int lane) {
  const unsigned short* p = base + (size_t)(row0 + (lane & 15)) * ld + k0 + ((lane & 16) ? 8 : 0);
  f.q[0] = *(const U4*)p;
  f.q[1] = *(const U4*)(p + 16);
}
// B fragment (32x16) from row-major [N,K] ("NT"): lane holds col n0+(lane&15),
// K {k0..k0+7, k0+16..k0+23} (lanes 0-15) / {k0+8..15, k0+24..31} (lanes 16-31).
DEV void load_frag_b(FragBF& f, const unsigned short* base, int col0, int k0, int ld, int lane) {
  const unsigned short* p = base + (size_t)(col0 + (lane & 15)) * ld + k0 + ((lane & 16) ? 8 : 0);
  f.q[0] = *(const U4*)p;
  f.q[1] = *(const U4*)(p + 16);
}

// ---------------------------------------------------------------- cvt
__global__ void cvt_f32_to_bf16(const float* __restrict__ in, unsigned short* __restrict__ out, int n) {
  for (int i = blockIdx.x * blockDim.x + threadIdx.x; i < n; i += gridDim.x * blockDim.x)
    out[i] = f2bf(in[i]);
}

// ---------------------------------------------------------------- GEMM (NT)
// C[M,N] = A[M,K] * W[N,K]^T + bias.  Block 256 = 8 waves (2M x 4N),
// block tile 128x128, wave tile 64x32 (4x2 WMMA tiles), K step 32.
// A/B 128x32 tiles staged to LDS by async b128 copies, double-buffered.
DEV void stage_tile(const unsigned short* g, int grow0, int gld, int kk,
                    unsigned short* s, int tid) {
#pragma unroll
  for (int j = 0; j < 2; ++j) {
    const int c = tid * 2 + j;            // 0..511 : 16B segment id
    const int r = c >> 2, seg = c & 3;    // row 0..127, 16B segment 0..3
    async_copy_b128(g + (size_t)(grow0 + r) * gld + kk + seg * 8, s + r * 32 + seg * 8);
  }
}

template <bool OUT_BF16>
__global__ __launch_bounds__(256) void gemm_nt(const unsigned short* __restrict__ A,
                                               const unsigned short* __restrict__ W,
                                               const float* __restrict__ bias,
                                               void* __restrict__ Cout,
                                               int M, int N, int Kd) {
  __shared__ __attribute__((aligned(16))) unsigned short As[2][128 * 32];
  __shared__ __attribute__((aligned(16))) unsigned short Bs[2][128 * 32];

  const int tid = threadIdx.x;
  const int lane = tid & 31;
  const int wave = tid >> 5;
  const int mloc = (wave & 1) * 64;             // local row in 128-tile
  const int nloc = (wave >> 1) * 32;            // local col in 128-tile
  const int m0 = blockIdx.y * 128 + mloc;
  const int n0 = blockIdx.x * 128 + nloc;

  v8f acc[4][2] = {};

  stage_tile(A, blockIdx.y * 128, Kd, 0, As[0], tid);
  stage_tile(W, blockIdx.x * 128, Kd, 0, Bs[0], tid);
  wait_async0();
  __syncthreads();

  int buf = 0;
  for (int kk = 0; kk < Kd; kk += 32) {
    if (kk + 32 < Kd) {                         // prefetch next K-slice
      stage_tile(A, blockIdx.y * 128, Kd, kk + 32, As[buf ^ 1], tid);
      stage_tile(W, blockIdx.x * 128, Kd, kk + 32, Bs[buf ^ 1], tid);
    }
    FragBF a[4], b[2];
#pragma unroll
    for (int mt = 0; mt < 4; ++mt) load_frag_a(a[mt], As[buf], mloc + 16 * mt, 0, 32, lane);
#pragma unroll
    for (int nt = 0; nt < 2; ++nt) load_frag_b(b[nt], Bs[buf], nloc + 16 * nt, 0, 32, lane);
#pragma unroll
    for (int mt = 0; mt < 4; ++mt)
#pragma unroll
      for (int nt = 0; nt < 2; ++nt)
        acc[mt][nt] = wmma_bf16(a[mt], b[nt], acc[mt][nt]);
    wait_async0();
    __syncthreads();
    buf ^= 1;
  }

#pragma unroll
  for (int mt = 0; mt < 4; ++mt)
#pragma unroll
    for (int nt = 0; nt < 2; ++nt) {
      const int col = n0 + 16 * nt + (lane & 15);
      const float bv = bias ? bias[col] : 0.0f;
      const int mb = m0 + 16 * mt + ((lane & 16) ? 8 : 0);
#pragma unroll
      for (int i = 0; i < 8; ++i) {
        const float v = acc[mt][nt][i] + bv;
        const size_t idx = (size_t)(mb + i) * N + col;
        if (OUT_BF16) ((unsigned short*)Cout)[idx] = f2bf(v);
        else          ((float*)Cout)[idx] = v;
      }
    }
}

// ---------------------------------------------------------------- beta proj
__global__ __launch_bounds__(128) void beta_proj(const float* __restrict__ x,
                                                 const float* __restrict__ Wb,
                                                 const float* __restrict__ bb,
                                                 float* __restrict__ beta) {
  __shared__ float xr[Dd];
  const int row = blockIdx.x;  // b*L + l
  for (int i = threadIdx.x; i < Dd; i += 128) xr[i] = x[(size_t)row * Dd + i];
  __syncthreads();
  const int h = threadIdx.x >> 3, part = threadIdx.x & 7;
  const float* wr = Wb + (size_t)h * Dd + part * 128;
  const float* xp = xr + part * 128;
  float s = 0.f;
#pragma unroll 8
  for (int j = 0; j < 128; ++j) s += xp[j] * wr[j];
  s += __shfl_xor(s, 1, 8);
  s += __shfl_xor(s, 2, 8);
  s += __shfl_xor(s, 4, 8);
  if (part == 0) {
    const float z = s + bb[h];
    const int b = row >> 11, l = row & (Ls - 1);
    beta[((size_t)b * Hh + h) * Ls + l] = 1.f / (1.f + __expf(-z));
  }
}

// ---------------------------------------------------------------- depthwise causal conv (K=4)
__global__ __launch_bounds__(256) void dwconv4(const unsigned short* __restrict__ in,
                                               const float* __restrict__ w,
                                               const float* __restrict__ bias,
                                               unsigned short* __restrict__ out) {
  const int total = BL * Dd;
  for (int idx = blockIdx.x * blockDim.x + threadIdx.x; idx < total; idx += gridDim.x * blockDim.x) {
    const int d = idx & (Dd - 1);
    const int l = (idx >> 10) & (Ls - 1);
    float acc = bias[d];
#pragma unroll
    for (int i = 0; i < 4; ++i) {
      const int ls = l + i - 3;
      if (ls >= 0) acc += w[d * 4 + i] * bf2f(in[idx + (i - 3) * Dd]);
    }
    out[idx] = f2bf(acc);
  }
}

// ---------------------------------------------------------------- per-head norms + relayout to [B,H,L,HD]
__global__ __launch_bounds__(128) void qkv_normalize(const unsigned short* __restrict__ q,
                                                     const unsigned short* __restrict__ k,
                                                     const unsigned short* __restrict__ v,
                                                     unsigned short* __restrict__ qh,
                                                     unsigned short* __restrict__ kh,
                                                     unsigned short* __restrict__ vh) {
  const int lane = threadIdx.x & 31;
  const int gw = blockIdx.x * 4 + (threadIdx.x >> 5);   // one wave per (b,l,h)
  const int h = gw & (Hh - 1);
  const int l = (gw >> 4) & (Ls - 1);
  const int b = gw >> 15;
  const size_t src = ((size_t)(b * Ls + l)) * Dd + h * HD;
  const unsigned int qu = ((const unsigned int*)(q + src))[lane];
  const unsigned int ku = ((const unsigned int*)(k + src))[lane];
  const unsigned int vu = ((const unsigned int*)(v + src))[lane];
  float q0 = bf2f(qu & 0xFFFF), q1 = bf2f(qu >> 16);
  float k0 = bf2f(ku & 0xFFFF), k1 = bf2f(ku >> 16);
  float v0 = bf2f(vu & 0xFFFF), v1 = bf2f(vu >> 16);
  v0 = v0 / (1.f + __expf(-v0));      // silu
  v1 = v1 / (1.f + __expf(-v1));
  float kss = k0 * k0 + k1 * k1;
  float qss = q0 * q0 + q1 * q1;
#pragma unroll
  for (int m = 1; m < 32; m <<= 1) { kss += __shfl_xor(kss, m, 32); qss += __shfl_xor(qss, m, 32); }
  const float kr = 1.f / (sqrtf(kss) + 1e-6f);
  const float qr = 1.f / (sqrtf(qss) + 1e-6f);
  const size_t dst = ((size_t)((b * Hh + h) * Ls + l)) * HD;
  ((unsigned int*)(qh + dst))[lane] = (unsigned)f2bf(q0 * qr) | ((unsigned)f2bf(q1 * qr) << 16);
  ((unsigned int*)(kh + dst))[lane] = (unsigned)f2bf(k0 * kr) | ((unsigned)f2bf(k1 * kr) << 16);
  ((unsigned int*)(vh + dst))[lane] = (unsigned)f2bf(v0 * kr) | ((unsigned)f2bf(v1 * kr) << 16);
}

// ---------------------------------------------------------------- chunked delta rule (WY form)
// One block per (b,h); 4 waves; state S (64x64) persistent in WMMA accumulators
// (16-row strip per wave); bf16 transposed mirror STb in LDS for B-operands.
__global__ __launch_bounds__(128) void delta_chunk(const unsigned short* __restrict__ qh,
                                                   const unsigned short* __restrict__ kh,
                                                   const unsigned short* __restrict__ vh,
                                                   const float* __restrict__ beta,
                                                   float* __restrict__ o) {
  extern __shared__ unsigned char smem[];
  unsigned short* Qb   = (unsigned short*)(smem);            // Q chunk, row-major [t][d]
  unsigned short* Kb   = (unsigned short*)(smem + 8192);     // K chunk [t][d]
  unsigned short* KbT  = (unsigned short*)(smem + 16384);    // K^T [d][t]
  unsigned short* KbbT = (unsigned short*)(smem + 24576);    // (beta*K)^T [d][t]
  unsigned short* VbT  = (unsigned short*)(smem + 32768);    // (beta*V)^T [d][t]
  unsigned short* Ab   = (unsigned short*)(smem + 40960);    // T matrix, later masked QK^T
  unsigned short* STb  = (unsigned short*)(smem + 49152);    // S^T bf16 [v][d]
  float*          attnf= (float*)(smem + 57344);             // 16KB, aliases Wnb+UiT
  unsigned short* Wnb  = (unsigned short*)(smem + 57344);    // -(T @ kbeta) [t][d]
  unsigned short* UiT  = (unsigned short*)(smem + 65536);    // u_i^T [v][t]
  float* rowbuf        = (float*)(smem + 73728);
  float* bet           = (float*)(smem + 73984);

  const int tid = threadIdx.x;
  const int lane = tid & 31;
  const int wave = tid >> 5;
  const int m0 = wave * 16;

  const size_t sbase = (size_t)blockIdx.x * Ls * HD;
  const unsigned short* qg = qh + sbase;
  const unsigned short* kg = kh + sbase;
  const unsigned short* vg = vh + sbase;
  const float* bg = beta + (size_t)blockIdx.x * Ls;
  float* og = o + sbase;

  v8f Sacc[4] = {};
  for (int e = tid; e < 4096; e += 128) STb[e] = 0;
  __syncthreads();

  for (int ch = 0; ch < NCHUNK; ++ch) {
    // ---- async load Q,K chunk (8KB each) + beta ----
    {
      const U4* qs = (const U4*)(qg + ch * 4096);
      const U4* ks = (const U4*)(kg + ch * 4096);
      for (int i = tid; i < 512; i += 128) {
        async_copy_b128(qs + i, ((U4*)Qb) + i);
        async_copy_b128(ks + i, ((U4*)Kb) + i);
      }
      if (tid < 64) bet[tid] = bg[ch * 64 + tid];
      wait_async0();
    }
    __syncthreads();
    // ---- transposed / beta-scaled operands ----
    for (int e = tid; e < 4096; e += 128) {
      const int r = e >> 6, c = e & 63;
      const float bv = bet[r];
      const unsigned short kv = Kb[e];
      KbT[c * 64 + r]  = kv;
      KbbT[c * 64 + r] = f2bf(bv * bf2f(kv));
      VbT[c * 64 + r]  = f2bf(bv * bf2f(vg[ch * 4096 + e]));
    }
    __syncthreads();
    // ---- M1: raw = K K^T ; attnf = -beta_i*raw strictly-lower, 0 elsewhere ----
    {
      v8f acc[4] = {};
      for (int ks = 0; ks < 64; ks += 32) {
        FragBF fa; load_frag_a(fa, Kb, m0, ks, 64, lane);
#pragma unroll
        for (int nt = 0; nt < 4; ++nt) {
          FragBF fb; load_frag_b(fb, Kb, nt * 16, ks, 64, lane);
          acc[nt] = wmma_bf16(fa, fb, acc[nt]);
        }
      }
#pragma unroll
      for (int nt = 0; nt < 4; ++nt) {
        const int n = nt * 16 + (lane & 15);
        const int mb = m0 + ((lane & 16) ? 8 : 0);
#pragma unroll
        for (int i = 0; i < 8; ++i) {
          const int m = mb + i;
          attnf[m * 64 + n] = (n < m) ? (-bet[m] * acc[nt][i]) : 0.0f;
        }
      }
    }
    __syncthreads();
    // ---- forward substitution: rows become (I + L)^-1 strictly-lower part ----
    for (int i = 1; i < 64; ++i) {
      if (tid < 64) rowbuf[tid] = (tid < i) ? attnf[i * 64 + tid] : 0.0f;
      __syncthreads();
      if (tid < i) {
        float s = rowbuf[tid];
        for (int j = 0; j < i; ++j) s += rowbuf[j] * attnf[j * 64 + tid];
        attnf[i * 64 + tid] = s;
      }
      __syncthreads();
    }
    if (tid < 64) attnf[tid * 64 + tid] = 1.0f;
    __syncthreads();
    for (int e = tid; e < 4096; e += 128) Ab[e] = f2bf(attnf[e]);
    __syncthreads();
    // ---- M2: Wnb = -(T @ kbeta)  (overwrites attnf region; attnf is dead) ----
    {
      v8f acc[4] = {};
      for (int ks = 0; ks < 64; ks += 32) {
        FragBF fa; load_frag_a(fa, Ab, m0, ks, 64, lane);
#pragma unroll
        for (int nt = 0; nt < 4; ++nt) {
          FragBF fb; load_frag_b(fb, KbbT, nt * 16, ks, 64, lane);
          acc[nt] = wmma_bf16(fa, fb, acc[nt]);
        }
      }
#pragma unroll
      for (int nt = 0; nt < 4; ++nt) {
        const int n = nt * 16 + (lane & 15);
        const int mb = m0 + ((lane & 16) ? 8 : 0);
#pragma unroll
        for (int i = 0; i < 8; ++i) Wnb[(mb + i) * 64 + n] = f2bf(-acc[nt][i]);
      }
    }
    __syncthreads();
    // ---- M3+M4: u_i = T @ vbeta + Wnb @ S ; store transposed ----
    {
      v8f acc[4] = {};
      for (int ks = 0; ks < 64; ks += 32) {
        FragBF fa; load_frag_a(fa, Ab, m0, ks, 64, lane);
#pragma unroll
        for (int nt = 0; nt < 4; ++nt) {
          FragBF fb; load_frag_b(fb, VbT, nt * 16, ks, 64, lane);
          acc[nt] = wmma_bf16(fa, fb, acc[nt]);
        }
      }
      for (int ks = 0; ks < 64; ks += 32) {
        FragBF fa; load_frag_a(fa, Wnb, m0, ks, 64, lane);
#pragma unroll
        for (int nt = 0; nt < 4; ++nt) {
          FragBF fb; load_frag_b(fb, STb, nt * 16, ks, 64, lane);
          acc[nt] = wmma_bf16(fa, fb, acc[nt]);
        }
      }
#pragma unroll
      for (int nt = 0; nt < 4; ++nt) {
        const int n = nt * 16 + (lane & 15);
        const int mb = m0 + ((lane & 16) ? 8 : 0);
#pragma unroll
        for (int i = 0; i < 8; ++i) UiT[n * 64 + (mb + i)] = f2bf(acc[nt][i]);
      }
    }
    __syncthreads();
    // ---- M5: o = Q @ S ; M6: masked QK^T -> Ab ----
    v8f oacc[4] = {};
    {
      for (int ks = 0; ks < 64; ks += 32) {
        FragBF fa; load_frag_a(fa, Qb, m0, ks, 64, lane);
#pragma unroll
        for (int nt = 0; nt < 4; ++nt) {
          FragBF fb; load_frag_b(fb, STb, nt * 16, ks, 64, lane);
          oacc[nt] = wmma_bf16(fa, fb, oacc[nt]);
        }
      }
      v8f qk[4] = {};
      for (int ks = 0; ks < 64; ks += 32) {
        FragBF fa; load_frag_a(fa, Qb, m0, ks, 64, lane);
#pragma unroll
        for (int nt = 0; nt < 4; ++nt) {
          FragBF fb; load_frag_b(fb, Kb, nt * 16, ks, 64, lane);
          qk[nt] = wmma_bf16(fa, fb, qk[nt]);
        }
      }
#pragma unroll
      for (int nt = 0; nt < 4; ++nt) {
        const int n = nt * 16 + (lane & 15);
        const int mb = m0 + ((lane & 16) ? 8 : 0);
#pragma unroll
        for (int i = 0; i < 8; ++i) {
          const int m = mb + i;
          Ab[m * 64 + n] = (n <= m) ? f2bf(qk[nt][i]) : (unsigned short)0;
        }
      }
    }
    __syncthreads();
    // ---- M7: o += masked(QK^T) @ u_i ; store ----
    {
      for (int ks = 0; ks < 64; ks += 32) {
        FragBF fa; load_frag_a(fa, Ab, m0, ks, 64, lane);
#pragma unroll
        for (int nt = 0; nt < 4; ++nt) {
          FragBF fb; load_frag_b(fb, UiT, nt * 16, ks, 64, lane);
          oacc[nt] = wmma_bf16(fa, fb, oacc[nt]);
        }
      }
#pragma unroll
      for (int nt = 0; nt < 4; ++nt) {
        const int n = nt * 16 + (lane & 15);
        const int mb = m0 + ((lane & 16) ? 8 : 0);
#pragma unroll
        for (int i = 0; i < 8; ++i) og[ch * 4096 + (mb + i) * 64 + n] = oacc[nt][i];
      }
    }
    // ---- M8: S += K^T @ u_i (register accumulators), rebuild STb ----
    for (int ks = 0; ks < 64; ks += 32) {
      FragBF fa; load_frag_a(fa, KbT, m0, ks, 64, lane);
#pragma unroll
      for (int nt = 0; nt < 4; ++nt) {
        FragBF fb; load_frag_b(fb, UiT, nt * 16, ks, 64, lane);
        Sacc[nt] = wmma_bf16(fa, fb, Sacc[nt]);
      }
    }
    __syncthreads();
#pragma unroll
    for (int nt = 0; nt < 4; ++nt) {
      const int n = nt * 16 + (lane & 15);       // v index
      const int mb = m0 + ((lane & 16) ? 8 : 0); // d index
#pragma unroll
      for (int i = 0; i < 8; ++i) STb[n * 64 + (mb + i)] = f2bf(Sacc[nt][i]);
    }
    __syncthreads();
  }
}

// ---------------------------------------------------------------- RMSNorm -> bf16 [B,L,D]
__global__ __launch_bounds__(128) void rmsnorm_to_bf16(const float* __restrict__ o,
                                                       const float* __restrict__ w,
                                                       unsigned short* __restrict__ on) {
  const int lane = threadIdx.x & 31;
  const int gw = blockIdx.x * 4 + (threadIdx.x >> 5);
  const int h = gw & (Hh - 1);
  const int l = (gw >> 4) & (Ls - 1);
  const int b = gw >> 15;
  const size_t src = ((size_t)((b * Hh + h) * Ls + l)) * HD + lane * 2;
  const float x0 = o[src], x1 = o[src + 1];
  float ss = x0 * x0 + x1 * x1;
#pragma unroll
  for (int m = 1; m < 32; m <<= 1) ss += __shfl_xor(ss, m, 32);
  const float r = rsqrtf(ss * (1.0f / 64.0f) + 1e-6f);
  const int d0 = lane * 2;
  const float y0 = x0 * r * w[d0], y1 = x1 * r * w[d0 + 1];
  const size_t dst = ((size_t)(b * Ls + l)) * Dd + h * HD + d0;
  ((unsigned int*)on)[dst >> 1] = (unsigned)f2bf(y0) | ((unsigned)f2bf(y1) << 16);
}

// ---------------------------------------------------------------- launch
extern "C" void kernel_launch(void* const* d_in, const int* in_sizes, int n_in,
                              void* d_out, int out_size, void* d_ws, size_t ws_size,
                              hipStream_t stream) {
  (void)in_sizes; (void)n_in; (void)out_size; (void)ws_size;
  const float* x      = (const float*)d_in[0];
  const float* Wq     = (const float*)d_in[1];
  const float* bq     = (const float*)d_in[2];
  const float* Wk     = (const float*)d_in[3];
  const float* bk     = (const float*)d_in[4];
  const float* Wv     = (const float*)d_in[5];
  const float* bv     = (const float*)d_in[6];
  const float* Wbeta  = (const float*)d_in[7];
  const float* bbeta  = (const float*)d_in[8];
  const float* Wo     = (const float*)d_in[9];
  const float* bo     = (const float*)d_in[10];
  const float* wq_c   = (const float*)d_in[11];
  const float* bq_c   = (const float*)d_in[12];
  const float* wk_c   = (const float*)d_in[13];
  const float* bk_c   = (const float*)d_in[14];
  const float* wv_c   = (const float*)d_in[15];
  const float* bv_c   = (const float*)d_in[16];
  const float* norm_w = (const float*)d_in[17];

  char* ws = (char*)d_ws;
  const size_t MB = 1ull << 20;
  unsigned short* xb    = (unsigned short*)(ws + 0);        // 32MB
  unsigned short* Wqb   = (unsigned short*)(ws + 32 * MB);  // 2MB each
  unsigned short* Wkb   = (unsigned short*)(ws + 34 * MB);
  unsigned short* Wvb   = (unsigned short*)(ws + 36 * MB);
  unsigned short* Wob   = (unsigned short*)(ws + 38 * MB);
  unsigned short* rawq  = (unsigned short*)(ws + 40 * MB);  // 32MB each
  unsigned short* rawk  = (unsigned short*)(ws + 72 * MB);
  unsigned short* rawv  = (unsigned short*)(ws + 104 * MB);
  unsigned short* postq = (unsigned short*)(ws + 136 * MB);
  unsigned short* postk = (unsigned short*)(ws + 168 * MB);
  unsigned short* postv = (unsigned short*)(ws + 200 * MB);
  float*          betab = (float*)(ws + 232 * MB);          // 1MB  -> peak 233MB
  // lifetime-based aliases:
  unsigned short* qhd = rawq;                               // raw dead after conv
  unsigned short* khd = rawk;
  unsigned short* vhd = rawv;
  float*          of  = (float*)(ws + 136 * MB);            // 64MB over postq+postk (dead)
  unsigned short* onb = postv;                              // 32MB over postv (dead)

  // 1) bf16 conversions
  cvt_f32_to_bf16<<<4096, 256, 0, stream>>>(x, xb, BL * Dd);
  cvt_f32_to_bf16<<<1024, 256, 0, stream>>>(Wq, Wqb, Dd * Dd);
  cvt_f32_to_bf16<<<1024, 256, 0, stream>>>(Wk, Wkb, Dd * Dd);
  cvt_f32_to_bf16<<<1024, 256, 0, stream>>>(Wv, Wvb, Dd * Dd);
  cvt_f32_to_bf16<<<1024, 256, 0, stream>>>(Wo, Wob, Dd * Dd);

  // 2) projection GEMMs (WMMA bf16 -> bf16, async double-buffered LDS)
  gemm_nt<true><<<dim3(8, 128), 256, 0, stream>>>(xb, Wqb, bq, rawq, BL, Dd, Dd);
  gemm_nt<true><<<dim3(8, 128), 256, 0, stream>>>(xb, Wkb, bk, rawk, BL, Dd, Dd);
  gemm_nt<true><<<dim3(8, 128), 256, 0, stream>>>(xb, Wvb, bv, rawv, BL, Dd, Dd);

  // 3) beta = sigmoid(x @ Wbeta^T + bbeta) -> [B,H,L]
  beta_proj<<<BL, 128, 0, stream>>>(x, Wbeta, bbeta, betab);

  // 4) depthwise causal conv
  dwconv4<<<8192, 256, 0, stream>>>(rawq, wq_c, bq_c, postq);
  dwconv4<<<8192, 256, 0, stream>>>(rawk, wk_c, bk_c, postk);
  dwconv4<<<8192, 256, 0, stream>>>(rawv, wv_c, bv_c, postv);

  // 5) silu + L2 norms + relayout to [B,H,L,HD]
  qkv_normalize<<<BL * Hh / 4, 128, 0, stream>>>(postq, postk, postv, qhd, khd, vhd);

  // 6) chunked delta rule (one block per (b,h)), 74240B dynamic LDS
  delta_chunk<<<Bb * Hh, 128, 74240, stream>>>(qhd, khd, vhd, betab, of);

  // 7) RMSNorm -> bf16 [B,L,D]
  rmsnorm_to_bf16<<<BL * Hh / 4, 128, 0, stream>>>(of, norm_w, onb);

  // 8) output GEMM (WMMA bf16 -> f32 out)
  gemm_nt<false><<<dim3(8, 128), 256, 0, stream>>>(onb, Wob, bo, d_out, BL, Dd, Dd);
}
